// CINN_37056977830494
// MI455X (gfx1250) — matrix operand order repaired and checked by reading, not compiled, not run on hardware
//
#include <hip/hip_runtime.h>
#include <hip/hip_bf16.h>

// ---------------------------------------------------------------------------
// Fused RQS coupling flow for MI455X (gfx1250, wave32, WMMA).
//   GEMM1 (B x 32) @ (32 x 256) -> relu -> h      [bf16 WMMA, f32 acc]
//   GEMM2 (B x 256) @ (256 x 928) -> theta (LDS only, never HBM)
//   RQS spline epilogue (f32 VALU), permuted store + logdet.
// ---------------------------------------------------------------------------

typedef __bf16 bf16_t;
typedef __attribute__((ext_vector_type(16))) __bf16 v16bf;
typedef __attribute__((ext_vector_type(8)))  __bf16 v8bf;
typedef __attribute__((ext_vector_type(8)))  float  v8f;

#define B_ROWS   131072
#define C_DIM    64
#define S1D      32
#define S2D      32
#define KBINS    10
#define HDIM     256
#define NOUT     928          // S2 * (3K-1) = 32*29
#define M_TILE   64
#define HALF_N   464          // 16 splines * 29 params
#define TH_STRIDE 467         // 464 + pad (odd -> fewer bank conflicts)
#define X_STRIDE  33
#define X1B_STRIDE 40         // 32 + 8 bf16 pad (stride 80B, conflict-free b128)
#define H_STRIDE  264         // 256 + 8 bf16 pad

#define BOUNDF  1.0f          // 0.2*log1p(exp(0.5*2*log(e^5-1))) == 0.2*5 == 1.0
#define MIN_BWF 0.001f
#define MIN_DF  0.001f

__device__ __forceinline__ float softplusf(float x) {
  return (x > 20.f) ? x : log1pf(expf(x));
}

__global__ void f32_to_bf16_kernel(const float* __restrict__ s,
                                   bf16_t* __restrict__ d, int n) {
  int i = blockIdx.x * 256 + threadIdx.x;
  if (i < n) d[i] = (bf16_t)s[i];
}

__global__ __launch_bounds__(256)
void flow_fused(const float* __restrict__ x,
                const bf16_t* __restrict__ W1b,   // [256][32] bf16 row-major
                const float* __restrict__ b1,
                const bf16_t* __restrict__ W2b,   // [928][256] bf16 row-major
                const float* __restrict__ b2,
                const float* __restrict__ wperm,  // [64][64] permutation
                float* __restrict__ z_out,        // [B][64]
                float* __restrict__ ld_out)       // [B]
{
  extern __shared__ char smem[];
  float* x1f   = (float*)smem;                     // [64][33] f32 x1 (for store)
  float* x2f   = x1f + M_TILE * X_STRIDE;          // [64][33] f32 x2
  float* y2    = x2f + M_TILE * X_STRIDE;          // [64][33] spline outputs
  float* theta = y2  + M_TILE * X_STRIDE;          // [16][467] staged theta half
  float* ladsc = theta + 16 * TH_STRIDE;           // [16][17] logdet partials
  float* lad   = ladsc + 16 * 17;                  // [64]
  int*   flags = (int*)(lad + M_TILE);             // [64] row-inside flags
  int*   perm  = flags + M_TILE;                   // [64]
  bf16_t* x1b  = (bf16_t*)(perm + M_TILE);         // [64][40] bf16 x1 (A frags)
  bf16_t* hbuf = x1b + M_TILE * X1B_STRIDE;        // [64][264] bf16 h

  const int t    = threadIdx.x;
  const int lane = t & 31;
  const int wave = t >> 5;
  const long row0 = (long)blockIdx.x * M_TILE;

  // ---- load x tile (coalesced), split x1/x2, make bf16 copy of x1 ----------
  for (int e = t; e < M_TILE * C_DIM; e += 256) {
    int m = e >> 6, c = e & 63;
    float v = x[(row0 + m) * C_DIM + c];
    if (c < S1D) { x1f[m * X_STRIDE + c] = v; x1b[m * X1B_STRIDE + c] = (bf16_t)v; }
    else         { x2f[m * X_STRIDE + (c - S1D)] = v; }
  }
  if (t < M_TILE) {
    lad[t] = 0.f;
    int p = 0;
    for (int j = 0; j < C_DIM; ++j)           // extract permutation from w_perm
      if (wperm[t * C_DIM + j] > 0.5f) p = j;
    perm[t] = p;
  }
  __syncthreads();

  // ---- per-row inside flag: all |x2| <= BOUND ------------------------------
  if (t < M_TILE) {
    int ok = 1;
    for (int c = 0; c < S2D; ++c) {
      float v = x2f[t * X_STRIDE + c];
      ok &= (v >= -BOUNDF) & (v <= BOUNDF);
    }
    flags[t] = ok;
  }

  // ---- GEMM1: h = relu(x1 @ W1^T + b1)  (64x256, K=32, one WMMA k-step) ----
  {
    const int rb     = wave & 3;              // row block 0..3
    const int ntbase = (wave >> 2) * 8;       // n-tile group 0..7 / 8..15
    const int r  = rb * 16 + (lane & 15);
    const int c0 = (lane >> 4) * 16;
    v16bf afrag;
    v8bf* ap = (v8bf*)&afrag;
    ap[0] = *(const v8bf*)&x1b[r * X1B_STRIDE + c0];
    ap[1] = *(const v8bf*)&x1b[r * X1B_STRIDE + c0 + 8];
    for (int nt = ntbase; nt < ntbase + 8; ++nt) {
      const int ncol = nt * 16 + (lane & 15);
      v16bf bfrag = *(const v16bf*)&W1b[ncol * S1D + (lane >> 4) * 16];
      v8f acc = {};
      acc = __builtin_amdgcn_wmma_f32_16x16x32_bf16(
              false, afrag, false, bfrag, (short)0, acc, false, false);
      const float bias = b1[ncol];
      const int mrow = rb * 16 + 8 * (lane >> 4);
      #pragma unroll
      for (int rr = 0; rr < 8; ++rr) {
        float v = acc[rr] + bias;
        v = v > 0.f ? v : 0.f;
        hbuf[(mrow + rr) * H_STRIDE + ncol] = (bf16_t)v;
      }
    }
  }
  __syncthreads();

  // ---- GEMM2 + spline, per 16-row block, N split in two 464-col halves -----
  for (int rb = 0; rb < 4; ++rb) {
    // preload the 8 A-fragments (h rows of this block) once per row block
    v16bf afr[8];
    {
      const int r = rb * 16 + (lane & 15);
      #pragma unroll
      for (int kk = 0; kk < 8; ++kk) {
        const int base = kk * 32 + (lane >> 4) * 16;
        v8bf* ap = (v8bf*)&afr[kk];
        ap[0] = *(const v8bf*)&hbuf[r * H_STRIDE + base];
        ap[1] = *(const v8bf*)&hbuf[r * H_STRIDE + base + 8];
      }
    }
    for (int hh = 0; hh < 2; ++hh) {
      // theta half: columns [hh*464, hh*464+464), 29 n-tiles over 8 waves
      for (int nt = wave; nt < 29; nt += 8) {
        const int ng = hh * HALF_N + nt * 16 + (lane & 15);   // global theta col
        const bf16_t* wrow = &W2b[ng * HDIM + (lane >> 4) * 16];
        __builtin_prefetch(wrow, 0, 1);
        v8f acc = {};
        #pragma unroll
        for (int kk = 0; kk < 8; ++kk) {
          v16bf bfr = *(const v16bf*)&wrow[kk * 32];
          acc = __builtin_amdgcn_wmma_f32_16x16x32_bf16(
                  false, afr[kk], false, bfr, (short)0, acc, false, false);
        }
        const float bias = b2[ng];
        const int mb = 8 * (lane >> 4);
        const int nl = nt * 16 + (lane & 15);
        #pragma unroll
        for (int rr = 0; rr < 8; ++rr)
          theta[(mb + rr) * TH_STRIDE + nl] = acc[rr] + bias;
      }
      __syncthreads();

      // ---- RQS spline: 16 rows x 16 splines = 256 threads, 1 each ---------
      {
        const int mloc = t & 15, jj = t >> 4;
        const int m = rb * 16 + mloc;
        const int j = hh * 16 + jj;
        const float* th = &theta[mloc * TH_STRIDE + jj * 29];
        float th_[29];
        #pragma unroll
        for (int q = 0; q < 29; ++q) th_[q] = th[q];

        // bin widths: softmax(uw) -> affine -> cumsum knots
        float mx = th_[0];
        #pragma unroll
        for (int q = 1; q < KBINS; ++q) mx = fmaxf(mx, th_[q]);
        float ew[KBINS]; float sw = 0.f;
        #pragma unroll
        for (int q = 0; q < KBINS; ++q) { ew[q] = expf(th_[q] - mx); sw += ew[q]; }
        const float invw = 1.f / sw;
        float cw[KBINS + 1];
        cw[0] = -BOUNDF;
        float a = 0.f;
        #pragma unroll
        for (int q = 0; q < KBINS; ++q) {
          a += MIN_BWF + (1.f - MIN_BWF * KBINS) * ew[q] * invw;
          cw[q + 1] = 2.f * BOUNDF * a - BOUNDF;
        }
        cw[KBINS] = BOUNDF;

        // bin heights
        float mh = th_[KBINS];
        #pragma unroll
        for (int q = 1; q < KBINS; ++q) mh = fmaxf(mh, th_[KBINS + q]);
        float eh[KBINS]; float sh = 0.f;
        #pragma unroll
        for (int q = 0; q < KBINS; ++q) { eh[q] = expf(th_[KBINS + q] - mh); sh += eh[q]; }
        const float invh = 1.f / sh;
        float ch[KBINS + 1];
        ch[0] = -BOUNDF;
        a = 0.f;
        #pragma unroll
        for (int q = 0; q < KBINS; ++q) {
          a += MIN_BWF + (1.f - MIN_BWF * KBINS) * eh[q] * invh;
          ch[q + 1] = 2.f * BOUNDF * a - BOUNDF;
        }
        ch[KBINS] = BOUNDF;

        // derivatives: boundary = MIN_D + softplus(DERIV_CONST) == 1 exactly
        float dv[KBINS + 1];
        dv[0] = 1.f; dv[KBINS] = 1.f;
        #pragma unroll
        for (int q = 0; q < KBINS - 1; ++q)
          dv[q + 1] = MIN_DF + softplusf(th_[2 * KBINS + q]);

        const float xraw = x2f[m * X_STRIDE + j];
        const float xin  = fminf(fmaxf(xraw, -BOUNDF), BOUNDF);

        int idx = -1;
        #pragma unroll
        for (int q = 0; q <= KBINS; ++q) {
          const float cws = cw[q] + (q == KBINS ? 1e-6f : 0.f);
          idx += (xin >= cws) ? 1 : 0;
        }
        idx = idx < 0 ? 0 : (idx > KBINS - 1 ? KBINS - 1 : idx);

        const float icw = cw[idx], ibw = cw[idx + 1] - cw[idx];
        const float ich = ch[idx], ihv = ch[idx + 1] - ch[idx];
        const float idelta = ihv / ibw;
        const float idr = dv[idx], idr1 = dv[idx + 1];
        const float tt  = (xin - icw) / ibw;
        const float t1  = tt * (1.f - tt);
        const float num = ihv * (idelta * tt * tt + idr * t1);
        const float den = idelta + (idr + idr1 - 2.f * idelta) * t1;
        const float outv = ich + num / den;
        const float om = 1.f - tt;
        const float dnum = idelta * idelta *
                           (idr1 * tt * tt + 2.f * idelta * t1 + idr * om * om);
        const float lterm = logf(dnum) - 2.f * logf(den);

        const int ok = flags[m];
        y2[m * X_STRIDE + j] = ok ? outv : xraw;
        ladsc[mloc * 17 + jj] = ok ? lterm : 0.f;
      }
      __syncthreads();
      if (t < 16) {                       // deterministic logdet reduction
        float s = 0.f;
        #pragma unroll
        for (int q = 0; q < 16; ++q) s += ladsc[t * 17 + q];
        lad[rb * 16 + t] += s;
      }
      __syncthreads();
    }
  }

  // ---- permuted store: z[:, i] = y[:, perm[i]] ------------------------------
  for (int e = t; e < M_TILE * C_DIM; e += 256) {
    const int m = e >> 6, i = e & 63;
    const int p = perm[i];
    const float v = (p < S1D) ? x1f[m * X_STRIDE + p]
                              : y2[m * X_STRIDE + (p - S1D)];
    z_out[(row0 + m) * C_DIM + i] = v;
  }
  if (t < M_TILE) ld_out[row0 + t] = lad[t];
}

extern "C" void kernel_launch(void* const* d_in, const int* in_sizes, int n_in,
                              void* d_out, int out_size, void* d_ws, size_t ws_size,
                              hipStream_t stream) {
  const float* x     = (const float*)d_in[0];
  const float* W1    = (const float*)d_in[1];
  const float* b1    = (const float*)d_in[2];
  const float* W2    = (const float*)d_in[3];
  const float* b2    = (const float*)d_in[4];
  const float* wperm = (const float*)d_in[5];
  float* out = (float*)d_out;

  bf16_t* W1b = (bf16_t*)d_ws;                          // 256*32*2 = 16 KB
  bf16_t* W2b = (bf16_t*)((char*)d_ws + 16384);         // 928*256*2 = 464 KB

  f32_to_bf16_kernel<<<(HDIM * S1D + 255) / 256, 256, 0, stream>>>(W1, W1b, HDIM * S1D);
  f32_to_bf16_kernel<<<(NOUT * HDIM + 255) / 256, 256, 0, stream>>>(W2, W2b, NOUT * HDIM);

  const size_t smem =
      (size_t)(3 * M_TILE * X_STRIDE + 16 * TH_STRIDE + 16 * 17 + M_TILE) * 4  // f32
      + (size_t)(2 * M_TILE) * 4                                               // int
      + (size_t)(M_TILE * X1B_STRIDE + M_TILE * H_STRIDE) * 2;                 // bf16

  flow_fused<<<B_ROWS / M_TILE, 256, smem, stream>>>(
      x, W1b, b1, W2b, b2, wperm, out, out + (size_t)B_ROWS * C_DIM);
}
